// FC_89094801588783
// MI455X (gfx1250) — matrix operand order repaired and checked by reading, not compiled, not run on hardware
//
#include <hip/hip_runtime.h>
#include <hip/hip_bf16.h>

typedef __attribute__((ext_vector_type(16))) __bf16 v16bf;
typedef __attribute__((ext_vector_type(8)))  float  v8f;

#define BATCH 16
#define NPTS  8192
#define FIN   256
#define FOUT  256
#define FINP  264     // padded LDS row stride (bf16 units) = 528 B -> conflict-free
#define EPSV  1e-5f

// workspace layout (bytes)
#define WS_WBF   0                        // 256*256 bf16 = 131072 B
#define WS_SUM   (131072)                 // 16*256 f32  = 16384 B
#define WS_SQ    (131072 + 16384)
#define WS_SCALE (131072 + 2*16384)
#define WS_SHIFT (131072 + 3*16384)

// ---- Kernel 1: convert W to bf16, zero stat accumulators -------------------
__global__ void prep_kernel(const float* __restrict__ W, __bf16* __restrict__ wbf,
                            float* __restrict__ gsum, float* __restrict__ gsq) {
    int i = blockIdx.x * 256 + threadIdx.x;
    if (blockIdx.x < 256) {
        wbf[i] = (__bf16)W[i];
    } else {
        int j = i - 256 * 256;
        if (j < BATCH * FOUT) { gsum[j] = 0.f; gsq[j] = 0.f; }
    }
}

// ---- Kernel 2: bf16 WMMA GEMM + bias + per-(batch,feature) stats -----------
// Block: 256 threads = 8 waves. Block tile = 128 rows x 256 features.
// W (bf16, bank-conflict-padded) staged once into LDS; each wave keeps its
// 16 rows' full-K A fragments in registers and retires one D tile at a time.
__global__ __launch_bounds__(256)
void gemm_kernel(const float* __restrict__ x, const __bf16* __restrict__ wbf,
                 const float* __restrict__ bias, float* __restrict__ y,
                 float* __restrict__ gsum, float* __restrict__ gsq) {
    __shared__ __align__(16) __bf16 sW[FOUT * FINP];   // 132 KB (<= 320 KB/WG on CDNA5)
    __shared__ float s_sum[FOUT];
    __shared__ float s_sq[FOUT];

    const int tid = threadIdx.x;

    // Cooperative stage: W bf16 row-major -> LDS with padded stride.
    {
        const float4* __restrict__ src = (const float4*)wbf;   // 16B = 8 bf16 chunks
        #pragma unroll
        for (int it = 0; it < (FOUT * FIN) / (8 * 256); ++it) { // 32 iters
            int i   = it * 256 + tid;
            int row = i >> 5;        // 32 chunks per 256-elem row
            int c   = i & 31;
            *(float4*)(sW + row * FINP + c * 8) = src[i];
        }
    }
    s_sum[tid] = 0.f;
    s_sq[tid]  = 0.f;

    const int wave = tid >> 5;
    const int lane = tid & 31;
    const int lm   = lane & 15;
    const int lh   = lane >> 4;

    const long row0  = (long)blockIdx.x * 128 + wave * 16;
    const int  batch = (int)(row0 >> 13);

    // A fragments: 16 rows x K=256 -> 8 fragments of 16x32 bf16 (64 VGPRs),
    // converted from f32 once, reused for all 16 output tiles.
    const float* __restrict__ xrow = x + (row0 + lm) * FIN;
    v16bf afrag[8];
#pragma unroll
    for (int k = 0; k < 8; ++k) {
        const float* pa = xrow + k * 32 + 16 * lh;
        float4 a0 = *(const float4*)(pa);
        float4 a1 = *(const float4*)(pa + 4);
        float4 a2 = *(const float4*)(pa + 8);
        float4 a3 = *(const float4*)(pa + 12);
        v16bf a;
        a[0]  = (__bf16)a0.x; a[1]  = (__bf16)a0.y; a[2]  = (__bf16)a0.z; a[3]  = (__bf16)a0.w;
        a[4]  = (__bf16)a1.x; a[5]  = (__bf16)a1.y; a[6]  = (__bf16)a1.z; a[7]  = (__bf16)a1.w;
        a[8]  = (__bf16)a2.x; a[9]  = (__bf16)a2.y; a[10] = (__bf16)a2.z; a[11] = (__bf16)a2.w;
        a[12] = (__bf16)a3.x; a[13] = (__bf16)a3.y; a[14] = (__bf16)a3.z; a[15] = (__bf16)a3.w;
        afrag[k] = a;
    }

    __syncthreads();   // LDS W ready

    const v8f zero = {};
    const long ybase = row0 + 8 * lh;

#pragma unroll
    for (int nt = 0; nt < 16; ++nt) {
        const int col = nt * 16 + lm;
        // B fragments come from LDS: lane (col lm, half lh) reads 16 contiguous
        // K values of row `col`; padded stride spreads the 16 lanes over all banks.
        const __bf16* __restrict__ brow = sW + col * FINP + 16 * lh;

        v8f acc = zero;
#pragma unroll
        for (int k = 0; k < 8; ++k) {
            v16bf bfrag = *(const v16bf*)(brow + k * 32);   // 2x ds_load_b128
            acc = __builtin_amdgcn_wmma_f32_16x16x32_bf16(
                false, afrag[k], false, bfrag, (short)0, acc, false, false);
        }

        // Epilogue for this tile: bias, store y, local stats.
        const float bv = bias[col];
        float psum = 0.f, psq = 0.f;
#pragma unroll
        for (int r = 0; r < 8; ++r) {
            float v = acc[r] + bv;
            psum += v;
            psq  += v * v;
            y[(ybase + r) * FOUT + col] = v;
        }
        atomicAdd(&s_sum[col], psum);   // ds_add_f32
        atomicAdd(&s_sq[col],  psq);
    }

    __syncthreads();
    atomicAdd(&gsum[batch * FOUT + tid], s_sum[tid]);  // global_atomic_add_f32
    atomicAdd(&gsq[batch * FOUT + tid],  s_sq[tid]);
}

// ---- Kernel 3: fold stats into per-(batch,feature) scale/shift -------------
__global__ void stats_kernel(const float* __restrict__ gsum, const float* __restrict__ gsq,
                             const float* __restrict__ gamma, const float* __restrict__ beta,
                             float* __restrict__ scale, float* __restrict__ shift) {
    int i = blockIdx.x * 256 + threadIdx.x;  // 16*256 entries
    int o = i & (FOUT - 1);
    float mean = gsum[i] * (1.0f / NPTS);
    float var  = gsq[i] * (1.0f / NPTS) - mean * mean;
    float rs   = rsqrtf(var + EPSV);
    float sc   = gamma[o] * rs;
    scale[i] = sc;
    shift[i] = beta[o] - mean * sc;
}

// ---- Kernel 4: in-place normalize, float4-vectorized -----------------------
__global__ void norm_kernel(float* __restrict__ y, const float* __restrict__ scale,
                            const float* __restrict__ shift) {
    long t   = (long)blockIdx.x * 256 + threadIdx.x;
    long row = t >> 6;                // 64 float4 per row
    int  c4  = (int)(t & 63);
    int  batch = (int)(row >> 13);
    float4* yp = (float4*)y + row * 64 + c4;
    const float4 s = ((const float4*)scale)[batch * 64 + c4];
    const float4 h = ((const float4*)shift)[batch * 64 + c4];
    float4 v = *yp;
    v.x = v.x * s.x + h.x;
    v.y = v.y * s.y + h.y;
    v.z = v.z * s.z + h.z;
    v.w = v.w * s.w + h.w;
    *yp = v;
}

extern "C" void kernel_launch(void* const* d_in, const int* in_sizes, int n_in,
                              void* d_out, int out_size, void* d_ws, size_t ws_size,
                              hipStream_t stream) {
    const float* x     = (const float*)d_in[0];
    const float* W     = (const float*)d_in[1];
    const float* bias  = (const float*)d_in[2];
    const float* gamma = (const float*)d_in[3];
    const float* beta  = (const float*)d_in[4];
    float* out = (float*)d_out;

    char* ws = (char*)d_ws;
    __bf16* wbf  = (__bf16*)(ws + WS_WBF);
    float* gsum  = (float*)(ws + WS_SUM);
    float* gsq   = (float*)(ws + WS_SQ);
    float* scale = (float*)(ws + WS_SCALE);
    float* shift = (float*)(ws + WS_SHIFT);

    // 256 blocks convert W (65536 elems), 16 blocks zero the 4096+4096 stat slots
    prep_kernel<<<272, 256, 0, stream>>>(W, wbf, gsum, gsq);

    // 131072 rows / 128 rows per block = 1024 blocks
    gemm_kernel<<<1024, 256, 0, stream>>>(x, wbf, bias, out, gsum, gsq);

    // 16*256 = 4096 scale/shift entries
    stats_kernel<<<16, 256, 0, stream>>>(gsum, gsq, gamma, beta, scale, shift);

    // 16*8192*64 float4 = 8388608 -> 32768 blocks
    norm_kernel<<<32768, 256, 0, stream>>>(out, scale, shift);
}
